// SocialGCNLayer_72945724555833
// MI455X (gfx1250) — matrix operand order repaired and checked by reading, not compiled, not run on hardware
//
#include <hip/hip_runtime.h>

#define N_NODES 50000
#define N_EDGES 800000
#define DFEAT   512

typedef _Float16 half_t;
typedef __attribute__((ext_vector_type(16))) _Float16 v16h;
typedef __attribute__((ext_vector_type(8)))  _Float16 v8h;
typedef __attribute__((ext_vector_type(4)))  _Float16 v4h;
typedef __attribute__((ext_vector_type(8)))  float    v8f;

// workspace layout (bytes)
#define WS_H_OFF   0ull                              // h: N_NODES*DFEAT f32 = 102,400,000 B
#define WS_A16_OFF 102400000ull                      // u_f in f16: 51,200,000 B
#define WS_W16_OFF (102400000ull + 51200000ull)      // W^T in f16: 524,288 B

// ---- convert u_f (f32) -> A16 (f16), 4 elems/thread, sizes divide exactly ----
__global__ __launch_bounds__(256) void gcn_cvt_a16(const float* __restrict__ u,
                                                   half_t* __restrict__ a16) {
  size_t i = ((size_t)blockIdx.x * blockDim.x + threadIdx.x) * 4;
  float4 f = *(const float4*)(u + i);
  v4h h;
  h[0] = (_Float16)f.x; h[1] = (_Float16)f.y;
  h[2] = (_Float16)f.z; h[3] = (_Float16)f.w;
  *(v4h*)(a16 + i) = h;
}

// ---- convert weight [k][n] (f32) -> W16T [n][k] (f16) ----
__global__ __launch_bounds__(256) void gcn_cvt_wt(const float* __restrict__ w,
                                                  half_t* __restrict__ wt) {
  int i = blockIdx.x * blockDim.x + threadIdx.x;  // 262144 threads
  int n = i >> 9;
  int k = i & 511;
  wt[i] = (_Float16)w[k * DFEAT + n];
}

// ---- WMMA GEMM: H[m][n] = sum_k A16[m][k] * W16T[n][k], f32 accumulate ----
// block = 8 waves; wave tile = 16(M) x 64(N); K loop in steps of 32.
__global__ __launch_bounds__(256) void gcn_gemm_wmma(const half_t* __restrict__ A16,
                                                     const half_t* __restrict__ WT16,
                                                     float* __restrict__ H) {
  const int lane = threadIdx.x & 31;
  const int wave = threadIdx.x >> 5;
  const int m0   = blockIdx.x * 128 + wave * 16;
  if (m0 >= N_NODES) return;                 // wave-uniform; EXEC stays all-1s
  const int n0 = blockIdx.y * 64;
  const int hi = lane >> 4;                  // lane half (ISA fragment layout)
  const int lm = lane & 15;

  int mrow = m0 + lm;
  if (mrow >= N_NODES) mrow = N_NODES - 1;   // clamp loads; stores guarded below

  const half_t* aRow  = A16  + (size_t)mrow * DFEAT;
  const half_t* bCol0 = WT16 + (size_t)(n0 +  0 + lm) * DFEAT + 16 * hi;
  const half_t* bCol1 = WT16 + (size_t)(n0 + 16 + lm) * DFEAT + 16 * hi;
  const half_t* bCol2 = WT16 + (size_t)(n0 + 32 + lm) * DFEAT + 16 * hi;
  const half_t* bCol3 = WT16 + (size_t)(n0 + 48 + lm) * DFEAT + 16 * hi;

  v8f acc0 = {}, acc1 = {}, acc2 = {}, acc3 = {};

  for (int k = 0; k < DFEAT; k += 32) {
    union { v16h v; v8h h[2]; } a, b0, b1, b2, b3;
    // A 16x32 layout (matches B): lanes<16 hold K 0..15, lanes>=16 hold K 16..31
    a.h[0] = *(const v8h*)(aRow + k + 16 * hi);
    a.h[1] = *(const v8h*)(aRow + k + 16 * hi + 8);
    // B 32x16 layout: lanes<16 hold K 0..15, lanes>=16 hold K 16..31 (N = lane&15)
    b0.h[0] = *(const v8h*)(bCol0 + k);  b0.h[1] = *(const v8h*)(bCol0 + k + 8);
    b1.h[0] = *(const v8h*)(bCol1 + k);  b1.h[1] = *(const v8h*)(bCol1 + k + 8);
    b2.h[0] = *(const v8h*)(bCol2 + k);  b2.h[1] = *(const v8h*)(bCol2 + k + 8);
    b3.h[0] = *(const v8h*)(bCol3 + k);  b3.h[1] = *(const v8h*)(bCol3 + k + 8);

    acc0 = __builtin_amdgcn_wmma_f32_16x16x32_f16(false, a.v, false, b0.v, (short)0, acc0, false, false);
    acc1 = __builtin_amdgcn_wmma_f32_16x16x32_f16(false, a.v, false, b1.v, (short)0, acc1, false, false);
    acc2 = __builtin_amdgcn_wmma_f32_16x16x32_f16(false, a.v, false, b2.v, (short)0, acc2, false, false);
    acc3 = __builtin_amdgcn_wmma_f32_16x16x32_f16(false, a.v, false, b3.v, (short)0, acc3, false, false);
  }

  // C/D layout: VGPR r holds M = r + 8*hi, N = lane&15
#pragma unroll
  for (int r = 0; r < 8; ++r) {
    int row = m0 + 8 * hi + r;
    if (row < N_NODES) {
      float* dst = H + (size_t)row * DFEAT + n0 + lm;
      dst[0]  = acc0[r];
      dst[16] = acc1[r];
      dst[32] = acc2[r];
      dst[48] = acc3[r];
    }
  }
}

// ---- zero output (deterministic init each call) ----
__global__ __launch_bounds__(256) void gcn_zero(float4* __restrict__ p) {
  size_t i = (size_t)blockIdx.x * blockDim.x + threadIdx.x;
  p[i] = make_float4(0.f, 0.f, 0.f, 0.f);
}

// ---- SpMM scatter with run-length accumulation (rows are sorted) ----
// block: 128 threads * float4 = all 512 columns; each block walks 128 edges.
__global__ __launch_bounds__(128) void gcn_spmm(const int* __restrict__ rows,
                                                const int* __restrict__ cols,
                                                const float* __restrict__ vals,
                                                const float* __restrict__ H,
                                                float* __restrict__ out) {
  const int e0    = blockIdx.x * 128;
  const int cbase = threadIdx.x * 4;
  const int eend  = (e0 + 128 < N_EDGES) ? (e0 + 128) : N_EDGES;

  float ax = 0.f, ay = 0.f, az = 0.f, aw = 0.f;
  int cur = -1;

  for (int e = e0; e < eend; ++e) {
    const int   r = rows[e];
    const int   c = cols[e];
    const float v = vals[e];
    const float4 hv = *(const float4*)(H + (size_t)c * DFEAT + cbase);
    if (r != cur) {                       // uniform branch across the block
      if (cur >= 0) {
        float* d = out + (size_t)cur * DFEAT + cbase;
        atomicAdd(d + 0, ax); atomicAdd(d + 1, ay);
        atomicAdd(d + 2, az); atomicAdd(d + 3, aw);
      }
      cur = r; ax = ay = az = aw = 0.f;
    }
    ax += v * hv.x; ay += v * hv.y; az += v * hv.z; aw += v * hv.w;
  }
  if (cur >= 0) {
    float* d = out + (size_t)cur * DFEAT + cbase;
    atomicAdd(d + 0, ax); atomicAdd(d + 1, ay);
    atomicAdd(d + 2, az); atomicAdd(d + 3, aw);
  }
}

// ---- ReLU in place ----
__global__ __launch_bounds__(256) void gcn_relu(float4* __restrict__ p) {
  size_t i = (size_t)blockIdx.x * blockDim.x + threadIdx.x;
  float4 v = p[i];
  v.x = v.x > 0.f ? v.x : 0.f;
  v.y = v.y > 0.f ? v.y : 0.f;
  v.z = v.z > 0.f ? v.z : 0.f;
  v.w = v.w > 0.f ? v.w : 0.f;
  p[i] = v;
}

extern "C" void kernel_launch(void* const* d_in, const int* in_sizes, int n_in,
                              void* d_out, int out_size, void* d_ws, size_t ws_size,
                              hipStream_t stream) {
  (void)in_sizes; (void)n_in; (void)out_size; (void)ws_size;

  const int*   adj_rows = (const int*)d_in[0];
  const int*   adj_cols = (const int*)d_in[1];
  const float* adj_vals = (const float*)d_in[2];
  const float* u_f      = (const float*)d_in[3];
  const float* weight   = (const float*)d_in[4];
  float* out = (float*)d_out;

  char* ws = (char*)d_ws;
  float*  H   = (float*)(ws + WS_H_OFF);
  half_t* A16 = (half_t*)(ws + WS_A16_OFF);
  half_t* W16 = (half_t*)(ws + WS_W16_OFF);

  // 25,600,000 elems / 4 per thread / 256 per block = 25000 blocks exactly
  gcn_cvt_a16<<<25000, 256, 0, stream>>>(u_f, A16);
  gcn_cvt_wt <<<1024,  256, 0, stream>>>(weight, W16);

  // M tiles: 50000/16 = 3125; 8 per block -> 391 blocks; N: 512/64 = 8
  gcn_gemm_wmma<<<dim3(391, 8), 256, 0, stream>>>(A16, W16, H);

  gcn_zero<<<25000, 256, 0, stream>>>((float4*)out);   // 6.4M float4
  gcn_spmm<<<6250, 128, 0, stream>>>(adj_rows, adj_cols, adj_vals, H, out);
  gcn_relu<<<25000, 256, 0, stream>>>((float4*)out);
}
